// CrossViewOTLoss_62775241998617
// MI455X (gfx1250) — compile-verified, matrix operand order checked by reading, not run on hardware
//
#include <hip/hip_runtime.h>
#include <hip/hip_bf16.h>
#include <math.h>

// ---------------------------------------------------------------------------
// Problem constants (match reference)
#define BB 2
#define CC 128
#define HH 128
#define NN 2048
#define NP1 2049              // N + dustbin row
#define TEMP_INV 10.0f        // 1 / 0.1
#define ITERS 20
#define EPSV 1e-8f

typedef __attribute__((ext_vector_type(16))) _Float16 v16h;
typedef __attribute__((ext_vector_type(8)))  _Float16 v8h;
typedef __attribute__((ext_vector_type(8)))  float    v8f;

// ---------------------------------------------------------------------------
// 0) zero the 3-float accumulator region in workspace
__global__ void ot_zero_accum(float* accum) {
    if (threadIdx.x < 4) accum[threadIdx.x] = 0.0f;
}

// ---------------------------------------------------------------------------
// 1) convert + transpose: qt[b][n][c] = (f16) feat[b][c][n], for both feats
__global__ void ot_transpose_f16(const float* __restrict__ feat_d,
                                 const float* __restrict__ feat_s,
                                 _Float16* __restrict__ qt_d,
                                 _Float16* __restrict__ qt_s) {
    size_t idx = (size_t)blockIdx.x * blockDim.x + threadIdx.x;
    const size_t total = (size_t)BB * CC * NN;
    if (idx >= total) return;
    int n = (int)(idx % NN);
    size_t t = idx / NN;
    int c = (int)(t % CC);
    int b = (int)(t / CC);
    size_t oidx = (((size_t)b * NN) + n) * CC + c;
    qt_d[oidx] = (_Float16)feat_d[idx];
    qt_s[oidx] = (_Float16)feat_s[idx];
}

// ---------------------------------------------------------------------------
// 2) dustbin head: dust[b][n] = b2 + sum_h W2[h]*relu(b1[h] + sum_c W1[h][c]*x[b][c][n])
//    Writes directly into row N (the appended dustbin row) of the log_T matrix.
__global__ void ot_dustbin(const float* __restrict__ feat_d,
                           const float* __restrict__ feat_s,
                           const float* __restrict__ W1,
                           const float* __restrict__ b1,
                           const float* __restrict__ W2,
                           const float* __restrict__ b2,
                           float* __restrict__ logT_ds,
                           float* __restrict__ logT_sd) {
    __shared__ float tile[CC * 128];   // 64 KB: feat tile [c][n_local]
    const float* feat = blockIdx.z ? feat_s : feat_d;
    float* out        = blockIdx.z ? logT_sd : logT_ds;
    int b   = blockIdx.y;
    int n0  = blockIdx.x * 128;
    int tid = threadIdx.x;             // 0..127
    for (int c = 0; c < CC; ++c)
        tile[c * 128 + tid] = feat[(((size_t)b * CC) + c) * NN + n0 + tid];
    __syncthreads();
    float dust = b2[0];
    for (int h = 0; h < HH; ++h) {
        float s = b1[h];
        const float* w = W1 + h * CC;
        #pragma unroll 8
        for (int c = 0; c < CC; ++c)
            s = fmaf(w[c], tile[c * 128 + tid], s);
        dust = fmaf(W2[h], fmaxf(s, 0.0f), dust);
    }
    out[(((size_t)b * NP1) + NN) * NN + n0 + tid] = dust;  // dustbin row, NOT /temp
}

// ---------------------------------------------------------------------------
// 3) score GEMM via WMMA f16 (f32 accumulate):
//    S[b][n][m] = sum_c feat_d[b][c][n]*feat_s[b][c][m]
//    logT_ds[b][n][m] = S/temp ; logT_sd[b][m][n] = S/temp (transposed reuse)
//    One wave computes a 16x64 strip (4 tiles, A reused 4x); K=128 in 4 steps.
//    Transposed stores staged through LDS for 64B-coalesced segments.
__global__ void __launch_bounds__(256)
ot_score_gemm(const _Float16* __restrict__ qt_d,   // [B][N][C]
              const _Float16* __restrict__ qt_s,   // [B][N][C]
              float* __restrict__ logT_ds,
              float* __restrict__ logT_sd) {
    __shared__ float tbuf[8][4][16][17];   // per-wave transpose staging, 34.8 KB

    int wave = threadIdx.x >> 5;           // 0..7
    int lane = threadIdx.x & 31;
    int b  = blockIdx.z;
    int n0 = blockIdx.x * 16;                    // A row-tile (query points)
    int m0 = (blockIdx.y * 8 + wave) * 64;       // this wave's 64-wide col strip

    int rowl = lane & 15;
    int hoff = (lane >> 4) * 8;            // per-lane K sub-offset (A/B layout)

    const _Float16* ab = qt_d + ((size_t)b * NN + (n0 + rowl)) * CC;
    const _Float16* bb = qt_s + ((size_t)b * NN + (m0 + rowl)) * CC;

    v8f acc[4] = {{}, {}, {}, {}};
    #pragma unroll
    for (int c0 = 0; c0 < CC; c0 += 32) {
        v8h a_lo = *(const v8h*)(ab + c0 + hoff);
        v8h a_hi = *(const v8h*)(ab + c0 + 16 + hoff);
        v16h A;
        #pragma unroll
        for (int i = 0; i < 8; ++i) { A[i] = a_lo[i]; A[i + 8] = a_hi[i]; }
        #pragma unroll
        for (int j = 0; j < 4; ++j) {
            const _Float16* bj = bb + (size_t)j * 16 * CC;
            v8h b_lo = *(const v8h*)(bj + c0 + hoff);
            v8h b_hi = *(const v8h*)(bj + c0 + 16 + hoff);
            v16h Bm;
            #pragma unroll
            for (int i = 0; i < 8; ++i) { Bm[i] = b_lo[i]; Bm[i + 8] = b_hi[i]; }
            acc[j] = __builtin_amdgcn_wmma_f32_16x16x32_f16(
                         false, A, false, Bm, (short)0, acc[j], false, false);
        }
    }

    int colm  = lane & 15;
    int rbase = (lane >> 4) * 8;

    // direct (row-major) store into logT_ds + stage transposed tiles in LDS
    #pragma unroll
    for (int j = 0; j < 4; ++j) {
        #pragma unroll
        for (int v = 0; v < 8; ++v) {
            int r = v + rbase;                 // n offset within tile
            float val = acc[j][v] * TEMP_INV;
            logT_ds[(((size_t)b * NP1) + (n0 + r)) * NN + (m0 + j * 16 + colm)] = val;
            tbuf[wave][j][colm][r] = val;      // transposed: T[m_local][n_local]
        }
    }
    __syncthreads();
    // coalesced transposed store: each half-wave writes one 64B row segment
    #pragma unroll
    for (int j = 0; j < 4; ++j) {
        #pragma unroll
        for (int v = 0; v < 8; ++v) {
            int mr = v + rbase;                // m offset within tile
            float tv = tbuf[wave][j][mr][colm];
            logT_sd[(((size_t)b * NP1) + (m0 + j * 16 + mr)) * NN + (n0 + colm)] = tv;
        }
    }
}

// ---------------------------------------------------------------------------
// 4a) row normalization: log_T -= logsumexp(log_T, axis=2). One block per row,
//     streaming (one-pass) LSE, row held in registers (8 f32/thread).
__global__ void __launch_bounds__(256)
ot_rownorm(float* __restrict__ logT_ds, float* __restrict__ logT_sd) {
    float* mat = blockIdx.y ? logT_sd : logT_ds;
    float* row = mat + (size_t)blockIdx.x * NN;   // blockIdx.x in [0, B*NP1)
    int tid = threadIdx.x;
    float x[8];
    float m = -INFINITY, s = 0.0f;
    #pragma unroll
    for (int i = 0; i < 8; ++i) {
        float xv = row[tid + i * 256];
        x[i] = xv;
        if (xv > m) { s = s * expf(m - xv) + 1.0f; m = xv; }
        else        { s += expf(xv - m); }
    }
    __shared__ float sm[256], ss[256];
    sm[tid] = m; ss[tid] = s;
    __syncthreads();
    for (int off = 128; off > 0; off >>= 1) {
        if (tid < off) {
            float m2 = sm[tid + off], s2 = ss[tid + off];
            float M = fmaxf(sm[tid], m2);
            ss[tid] = ss[tid] * expf(sm[tid] - M) + s2 * expf(m2 - M);
            sm[tid] = M;
        }
        __syncthreads();
    }
    float lse = sm[0] + logf(ss[0]);
    #pragma unroll
    for (int i = 0; i < 8; ++i)
        row[tid + i * 256] = x[i] - lse;
}

// 4b) column normalization: log_T -= logsumexp(log_T, axis=1).
//     One thread per column (coalesced across lanes), two passes (L2-resident).
__global__ void __launch_bounds__(256)
ot_colnorm(float* __restrict__ logT_ds, float* __restrict__ logT_sd) {
    float* mat = blockIdx.z ? logT_sd : logT_ds;
    int b = blockIdx.y;
    int mcol = blockIdx.x * 256 + threadIdx.x;
    float* base = mat + (size_t)b * NP1 * NN + mcol;
    float mx = -INFINITY, s = 0.0f;
    for (int n = 0; n < NP1; ++n) {
        float xv = base[(size_t)n * NN];
        if (xv > mx) { s = s * expf(mx - xv) + 1.0f; mx = xv; }
        else         { s += expf(xv - mx); }
    }
    float lse = mx + logf(s);
    for (int n = 0; n < NP1; ++n)
        base[(size_t)n * NN] -= lse;
}

// ---------------------------------------------------------------------------
// 5) loss reduction over the non-dustbin [B,N,N] region, 64x64 tiles.
//    T_sd^T fetched coalesced and transposed through LDS (65-float pitch).
//    accum[0] += (T_ds - T_sd^T)^2 ; accum[1] += ent_ds ; accum[2] += ent_sd
__global__ void __launch_bounds__(256)
ot_loss_reduce(const float* __restrict__ logT_ds,
               const float* __restrict__ logT_sd,
               float* __restrict__ accum) {
    __shared__ float sd_t[64][65];     // 16.6 KB
    int b  = blockIdx.z;
    int n0 = blockIdx.x * 64;
    int m0 = blockIdx.y * 64;
    int tx = threadIdx.x & 63;         // column within tile
    int ty = threadIdx.x >> 6;         // 0..3

    // load the [m0..m0+63][n0..n0+63] block of logT_sd, coalesced
    #pragma unroll 4
    for (int rr = 0; rr < 16; ++rr) {
        int r = rr * 4 + ty;
        sd_t[r][tx] = logT_sd[(((size_t)b * NP1) + (m0 + r)) * NN + (n0 + tx)];
    }
    __syncthreads();

    float c = 0.0f, e1 = 0.0f, e2 = 0.0f;
    #pragma unroll 4
    for (int rr = 0; rr < 16; ++rr) {
        int r = rr * 4 + ty;           // n offset within tile
        float t1 = expf(logT_ds[(((size_t)b * NP1) + (n0 + r)) * NN + (m0 + tx)]);
        float t2 = expf(sd_t[tx][r]);  // T_sd[m0+tx][n0+r] == T_sd^T element
        float d = t1 - t2;
        c  += d * d;
        e1 += -t1 * logf(t1 + EPSV);
        e2 += -t2 * logf(t2 + EPSV);
    }

    __shared__ float rc[256], r1[256], r2[256];
    int tid = threadIdx.x;
    rc[tid] = c; r1[tid] = e1; r2[tid] = e2;
    __syncthreads();
    for (int off = 128; off > 0; off >>= 1) {
        if (tid < off) {
            rc[tid] += rc[tid + off];
            r1[tid] += r1[tid + off];
            r2[tid] += r2[tid + off];
        }
        __syncthreads();
    }
    if (tid == 0) {
        atomicAdd(&accum[0], rc[0]);
        atomicAdd(&accum[1], r1[0]);
        atomicAdd(&accum[2], r2[0]);
    }
}

// 6) finalize scalar
__global__ void ot_finalize(const float* __restrict__ accum, float* __restrict__ out) {
    if (threadIdx.x == 0) {
        float cons = accum[0] / ((float)BB * (float)NN * (float)NN);
        float ent  = 0.5f * (accum[1] + accum[2]) / (float)BB;
        out[0] = cons + 0.01f * ent;
    }
}

// ---------------------------------------------------------------------------
extern "C" void kernel_launch(void* const* d_in, const int* in_sizes, int n_in,
                              void* d_out, int out_size, void* d_ws, size_t ws_size,
                              hipStream_t stream) {
    const float* feat_d = (const float*)d_in[0];
    const float* feat_s = (const float*)d_in[1];
    const float* W1     = (const float*)d_in[2];
    const float* b1     = (const float*)d_in[3];
    const float* W2     = (const float*)d_in[4];
    const float* b2     = (const float*)d_in[5];
    float* out = (float*)d_out;

    // workspace layout
    const size_t LOGT_ELEMS = (size_t)BB * NP1 * NN;      // 8,392,704 f32
    const size_t QT_ELEMS   = (size_t)BB * NN * CC;       //   524,288 f16
    float*    logT_ds = (float*)d_ws;
    float*    logT_sd = logT_ds + LOGT_ELEMS;
    _Float16* qt_d    = (_Float16*)(logT_sd + LOGT_ELEMS);
    _Float16* qt_s    = qt_d + QT_ELEMS;
    float*    accum   = (float*)(qt_s + QT_ELEMS);

    // 0) zero accumulators
    ot_zero_accum<<<1, 32, 0, stream>>>(accum);

    // 1) f16 transpose of both feature maps
    {
        size_t total = (size_t)BB * CC * NN;
        int blocks = (int)((total + 255) / 256);
        ot_transpose_f16<<<blocks, 256, 0, stream>>>(feat_d, feat_s, qt_d, qt_s);
    }

    // 2) dustbin head -> last row of each log_T
    ot_dustbin<<<dim3(NN / 128, BB, 2), 128, 0, stream>>>(
        feat_d, feat_s, W1, b1, W2, b2, logT_ds, logT_sd);

    // 3) WMMA score GEMM -> rows 0..N-1 of both log_T matrices
    //    grid: 128 n-tiles x (2048 / (8 waves * 64 cols)) = 4 col groups x B
    ot_score_gemm<<<dim3(NN / 16, NN / (8 * 64), BB), 256, 0, stream>>>(
        qt_d, qt_s, logT_ds, logT_sd);

    // 4) Sinkhorn iterations (row norm then col norm), both matrices fused
    for (int it = 0; it < ITERS; ++it) {
        ot_rownorm<<<dim3(BB * NP1, 2), 256, 0, stream>>>(logT_ds, logT_sd);
        ot_colnorm<<<dim3(NN / 256, BB, 2), 256, 0, stream>>>(logT_ds, logT_sd);
    }

    // 5) loss reduction (64x64 tiles) + 6) finalize
    ot_loss_reduce<<<dim3(NN / 64, NN / 64, BB), 256, 0, stream>>>(
        logT_ds, logT_sd, accum);
    ot_finalize<<<1, 32, 0, stream>>>(accum, out);
}
